// AttentionLayer_4741643895735
// MI455X (gfx1250) — matrix operand order repaired and checked by reading, not compiled, not run on hardware
//
#include <hip/hip_runtime.h>
#include <hip/hip_bf16.h>
#include <math.h>

#define D_MODEL 512
#define N_NODES 20
#define N_HEADS 8
#define DK 64
#define KCONV 1536           // 3 * D_MODEL (halo-concatenated conv K dim)
#define WSTRIDE (D_MODEL * KCONV)

// LDS layout (138240 B total, aliased)
#define XPAD_OFF 0            // xpad [23][512] f16 (rows 0,21,22 zero) ; later ao16 [20][512]
#define Q_OFF    23552        // q16  [20][512] f16
#define K_OFF    44032        // k16  [20][512] f16
#define VT_OFF   64512        // vT   [512][32] f16 (cols >=20 zero)
#define C_OFF    97280        // c32  [20][512] f32 ; aliased by s32 [8][20][32] f32
#define A16_OFF  117760       // a16  [8][32][32] f16 (inside c32 region tail)
#define SMEM_BYTES 138240

typedef __attribute__((ext_vector_type(16))) _Float16 v16h;
typedef __attribute__((ext_vector_type(8)))  _Float16 h8;
typedef __attribute__((ext_vector_type(8)))  float    v8f;

union F16F { v16h v; h8 h[2]; };

static __device__ __forceinline__ h8 ld8(const _Float16* p) { return *(const h8*)p; }

static __device__ __forceinline__ v8f wmma_f16(v16h a, v16h b, v8f c) {
    // D = A(16x32 f16) * B(32x16 f16) + C(16x16 f32)
    return __builtin_amdgcn_wmma_f32_16x16x32_f16(false, a, false, b, (short)0, c,
                                                  false, false);
}

// Branch-free A-fragment chunk for the halo-concatenated conv GEMM.
// kk = k*512 + i ; element = xpad[node + k][i]; invalid rows read zero row 22.
static __device__ __forceinline__ h8 xaug_chunk(const _Float16* xpad, int node, int kk) {
    int k = kk >> 9;
    int i = kk & 511;
    int row = (node < N_NODES) ? (node + k) : 22;   // v_cndmask, no exec games
    return ld8(xpad + row * D_MODEL + i);
}

// -------------------------------------------------------------------------
// One-shot weight conversion: f32 -> f16, conv weights transposed so that
// Wc[o][k*512+i] = W[o][i][k]  (B-fragment reads become contiguous 16B).
// -------------------------------------------------------------------------
__global__ void convert_weights(const float* __restrict__ Wq,
                                const float* __restrict__ Wk,
                                const float* __restrict__ Wv,
                                const float* __restrict__ Wo,
                                _Float16* __restrict__ Wcq,
                                _Float16* __restrict__ Wck,
                                _Float16* __restrict__ Wcv,
                                _Float16* __restrict__ Woc) {
    int idx = blockIdx.x * blockDim.x + threadIdx.x;
    int nthr = gridDim.x * blockDim.x;
    const int totalC = D_MODEL * KCONV;
    for (int t = idx; t < totalC; t += nthr) {
        int o = t / KCONV;
        int rem = t - o * KCONV;
        int k = rem >> 9;
        int i = rem & 511;
        int src = o * KCONV + i * 3 + k;
        Wcq[t] = (_Float16)Wq[src];
        Wck[t] = (_Float16)Wk[src];
        Wcv[t] = (_Float16)Wv[src];
    }
    for (int t = idx; t < D_MODEL * D_MODEL; t += nthr)
        Woc[t] = (_Float16)Wo[t];
}

// -------------------------------------------------------------------------
// Fused layer: conv-proj(QKV) + LN + residual + MHA(rel bias) + out proj.
// One workgroup (8 waves) per batch element.
// -------------------------------------------------------------------------
__global__ __launch_bounds__(256)
void attn_layer_kernel(const float* __restrict__ x,
                       const _Float16* __restrict__ Wcat,   // Wcq|Wck|Wcv consecutive
                       const float* __restrict__ lnq_g, const float* __restrict__ lnq_b,
                       const float* __restrict__ lnk_g, const float* __restrict__ lnk_b,
                       const float* __restrict__ lnv_g, const float* __restrict__ lnv_b,
                       const float* __restrict__ rel_table,
                       const _Float16* __restrict__ Woc,
                       const float* __restrict__ bo,
                       float* __restrict__ out) {
    __shared__ __align__(16) char smem[SMEM_BYTES];
    _Float16* xpad = (_Float16*)(smem + XPAD_OFF);
    _Float16* q16  = (_Float16*)(smem + Q_OFF);
    _Float16* k16  = (_Float16*)(smem + K_OFF);
    _Float16* vT   = (_Float16*)(smem + VT_OFF);
    float*    c32  = (float*)(smem + C_OFF);
    float*    s32  = (float*)(smem + C_OFF);
    _Float16* a16  = (_Float16*)(smem + A16_OFF);
    _Float16* ao16 = (_Float16*)(smem + XPAD_OFF);

    const int b    = blockIdx.x;
    const int tid  = threadIdx.x;
    const int wave = tid >> 5;
    const int lane = tid & 31;
    const int r16  = lane & 15;
    const int hi   = lane >> 4;

    // ---- stage 0: padded x image (f16) + zero vT --------------------------
    const float* xb = x + (size_t)b * (N_NODES * D_MODEL);
    for (int i = tid; i < 23 * D_MODEL; i += 256) {
        int row = i >> 9;
        int col = i & 511;
        float v = (row >= 1 && row <= N_NODES) ? xb[(row - 1) * D_MODEL + col] : 0.0f;
        xpad[i] = (_Float16)v;
    }
    for (int i = tid; i < D_MODEL * 32; i += 256) vT[i] = (_Float16)0.0f;
    __syncthreads();

    // ---- stage 1: Q/K/V conv-projections as [20x1536]@[1536x512] GEMMs ---
    _Float16* qkdst[2] = {q16, k16};

    for (int p = 0; p < 3; ++p) {
        const _Float16* W = Wcat + (size_t)p * WSTRIDE;   // stays addrspace(global)
        v8f acc[2][4] = {};
        for (int ks = 0; ks < 48; ++ks) {
            const int kkA0 = ks * 32 + hi * 8;        // A chunks: hi*8 , 16+hi*8
            F16F a0, a1;
            a0.h[0] = xaug_chunk(xpad, r16,      kkA0);
            a0.h[1] = xaug_chunk(xpad, r16,      kkA0 + 16);
            a1.h[0] = xaug_chunk(xpad, r16 + 16, kkA0);
            a1.h[1] = xaug_chunk(xpad, r16 + 16, kkA0 + 16);
            const int kkB0 = ks * 32 + hi * 8;        // B chunks: hi*8 , 16+hi*8
#pragma unroll
            for (int j = 0; j < 4; ++j) {
                const int o = (wave * 4 + j) * 16 + r16;
                const _Float16* bp = W + (size_t)o * KCONV + kkB0;
                __builtin_prefetch((const void*)(bp + 32), 0, 0); // next K step
                F16F bf;
                bf.h[0] = ld8(bp);
                bf.h[1] = ld8(bp + 16);
                acc[0][j] = wmma_f16(a0.v, bf.v, acc[0][j]);
                acc[1][j] = wmma_f16(a1.v, bf.v, acc[1][j]);
            }
        }
        // dump accumulators (valid rows only) for cross-wave LayerNorm
#pragma unroll
        for (int mt = 0; mt < 2; ++mt)
#pragma unroll
            for (int j = 0; j < 4; ++j)
#pragma unroll
                for (int e = 0; e < 8; ++e) {
                    int m = mt * 16 + hi * 8 + e;
                    if (m < N_NODES)
                        c32[m * D_MODEL + (wave * 4 + j) * 16 + r16] = acc[mt][j][e];
                }
        __syncthreads();

        // LayerNorm over D + residual; V goes to LDS transposed
        const float* g  = (p == 0) ? lnq_g : (p == 1) ? lnk_g : lnv_g;
        const float* bb = (p == 0) ? lnq_b : (p == 1) ? lnk_b : lnv_b;
        for (int row = wave; row < N_NODES; row += 8) {
            float s = 0.f, ss = 0.f;
            for (int d = lane; d < D_MODEL; d += 32) {
                float c = c32[row * D_MODEL + d];
                s += c; ss += c * c;
            }
#pragma unroll
            for (int m = 16; m >= 1; m >>= 1) {
                s  += __shfl_xor(s, m, 32);
                ss += __shfl_xor(ss, m, 32);
            }
            float mu = s * (1.0f / 512.0f);
            float rs = rsqrtf(ss * (1.0f / 512.0f) - mu * mu + 1e-5f);
            const float* xr = xb + row * D_MODEL;
            for (int d = lane; d < D_MODEL; d += 32) {
                float val = xr[d] + (c32[row * D_MODEL + d] - mu) * rs * g[d] + bb[d];
                if (p == 2) vT[d * 32 + row] = (_Float16)val;
                else        qkdst[p][row * D_MODEL + d] = (_Float16)val;
            }
        }
        __syncthreads();
    }

    // ---- stage 2: attention; wave w == head w -----------------------------
    {
        v8f sacc[2][2] = {};
#pragma unroll
        for (int ksn = 0; ksn < 2; ++ksn) {
            const int d0 = wave * DK + ksn * 32;
            F16F aq[2], bk[2];
#pragma unroll
            for (int mt = 0; mt < 2; ++mt) {
                int rr = mt * 16 + r16;
                if (rr < N_NODES) {
                    aq[mt].h[0] = ld8(q16 + rr * D_MODEL + d0 + hi * 8);
                    aq[mt].h[1] = ld8(q16 + rr * D_MODEL + d0 + 16 + hi * 8);
                } else { h8 z = {}; aq[mt].h[0] = z; aq[mt].h[1] = z; }
            }
#pragma unroll
            for (int nt = 0; nt < 2; ++nt) {
                int nn = nt * 16 + r16;
                if (nn < N_NODES) {
                    bk[nt].h[0] = ld8(k16 + nn * D_MODEL + d0 + hi * 8);
                    bk[nt].h[1] = ld8(k16 + nn * D_MODEL + d0 + 16 + hi * 8);
                } else { h8 z = {}; bk[nt].h[0] = z; bk[nt].h[1] = z; }
            }
#pragma unroll
            for (int mt = 0; mt < 2; ++mt)
#pragma unroll
                for (int nt = 0; nt < 2; ++nt)
                    sacc[mt][nt] = wmma_f16(aq[mt].v, bk[nt].v, sacc[mt][nt]);
        }
        const float scale = 0.125f;   // 1/sqrt(64)
#pragma unroll
        for (int mt = 0; mt < 2; ++mt)
#pragma unroll
            for (int nt = 0; nt < 2; ++nt)
#pragma unroll
                for (int e = 0; e < 8; ++e) {
                    int m = mt * 16 + hi * 8 + e;
                    if (m < N_NODES)
                        s32[(wave * N_NODES + m) * 32 + nt * 16 + r16] =
                            sacc[mt][nt][e] * scale;
                }
    }
    __syncthreads();

    // softmax with relative-position bias: lane = row
    {
        const int r = lane;
        _Float16* arow = a16 + (wave * 32 + r) * 32;
        if (r < N_NODES) {
            float vals[N_NODES];
            float mx = -1e30f;
#pragma unroll
            for (int j = 0; j < N_NODES; ++j) {
                float sv = s32[(wave * N_NODES + r) * 32 + j] +
                           rel_table[(r - j + N_NODES - 1) * N_HEADS + wave];
                vals[j] = sv;
                mx = fmaxf(mx, sv);
            }
            float sum = 0.f;
#pragma unroll
            for (int j = 0; j < N_NODES; ++j) {
                vals[j] = __expf(vals[j] - mx);
                sum += vals[j];
            }
            float inv = 1.0f / sum;
#pragma unroll
            for (int j = 0; j < N_NODES; ++j) arow[j] = (_Float16)(vals[j] * inv);
#pragma unroll
            for (int j = N_NODES; j < 32; ++j) arow[j] = (_Float16)0.0f;
        } else {
#pragma unroll
            for (int j = 0; j < 32; ++j) arow[j] = (_Float16)0.0f;
        }
    }
    __syncthreads();

    // P @ V  ->  ao16 (aliases dead xpad region)
    {
        v8f oacc[2][4] = {};
        F16F ap[2];
#pragma unroll
        for (int mt = 0; mt < 2; ++mt) {
            const _Float16* pA = a16 + (wave * 32 + mt * 16 + r16) * 32;
            ap[mt].h[0] = ld8(pA + hi * 8);
            ap[mt].h[1] = ld8(pA + 16 + hi * 8);
        }
#pragma unroll
        for (int nt = 0; nt < 4; ++nt) {
            const int d = wave * DK + nt * 16 + r16;
            F16F bv;
            bv.h[0] = ld8(vT + d * 32 + hi * 8);
            bv.h[1] = ld8(vT + d * 32 + 16 + hi * 8);
            oacc[0][nt] = wmma_f16(ap[0].v, bv.v, oacc[0][nt]);
            oacc[1][nt] = wmma_f16(ap[1].v, bv.v, oacc[1][nt]);
        }
#pragma unroll
        for (int mt = 0; mt < 2; ++mt)
#pragma unroll
            for (int nt = 0; nt < 4; ++nt)
#pragma unroll
                for (int e = 0; e < 8; ++e) {
                    int m = mt * 16 + hi * 8 + e;
                    if (m < N_NODES)
                        ao16[m * D_MODEL + wave * DK + nt * 16 + r16] =
                            (_Float16)oacc[mt][nt][e];
                }
    }
    __syncthreads();

    // ---- stage 3: output projection [20x512]@[512x512] + bias ------------
    {
        v8f yacc[2][4] = {};
        for (int ks = 0; ks < 16; ++ks) {
            const int kkA0 = ks * 32 + hi * 8;
            F16F a0, a1;
            a0.h[0] = ld8(ao16 + r16 * D_MODEL + kkA0);
            a0.h[1] = ld8(ao16 + r16 * D_MODEL + kkA0 + 16);
            if (16 + r16 < N_NODES) {
                a1.h[0] = ld8(ao16 + (16 + r16) * D_MODEL + kkA0);
                a1.h[1] = ld8(ao16 + (16 + r16) * D_MODEL + kkA0 + 16);
            } else { h8 z = {}; a1.h[0] = z; a1.h[1] = z; }
            const int kkB0 = ks * 32 + hi * 8;
#pragma unroll
            for (int j = 0; j < 4; ++j) {
                const int o = (wave * 4 + j) * 16 + r16;
                const _Float16* bp = Woc + (size_t)o * D_MODEL + kkB0;
                F16F bf;
                bf.h[0] = ld8(bp);
                bf.h[1] = ld8(bp + 16);
                yacc[0][j] = wmma_f16(a0.v, bf.v, yacc[0][j]);
                yacc[1][j] = wmma_f16(a1.v, bf.v, yacc[1][j]);
            }
        }
        float* ob = out + (size_t)b * (N_NODES * D_MODEL);
#pragma unroll
        for (int mt = 0; mt < 2; ++mt)
#pragma unroll
            for (int j = 0; j < 4; ++j)
#pragma unroll
                for (int e = 0; e < 8; ++e) {
                    int m = mt * 16 + hi * 8 + e;
                    if (m < N_NODES) {
                        int o = (wave * 4 + j) * 16 + r16;
                        ob[m * D_MODEL + o] = yacc[mt][j][e] + bo[o];
                    }
                }
    }
}

extern "C" void kernel_launch(void* const* d_in, const int* in_sizes, int n_in,
                              void* d_out, int out_size, void* d_ws, size_t ws_size,
                              hipStream_t stream) {
    const float* x      = (const float*)d_in[0];
    const float* Wq     = (const float*)d_in[1];
    const float* Wk     = (const float*)d_in[2];
    const float* Wv     = (const float*)d_in[3];
    const float* lnq_g  = (const float*)d_in[4];
    const float* lnq_b  = (const float*)d_in[5];
    const float* lnk_g  = (const float*)d_in[6];
    const float* lnk_b  = (const float*)d_in[7];
    const float* lnv_g  = (const float*)d_in[8];
    const float* lnv_b  = (const float*)d_in[9];
    const float* rel_t  = (const float*)d_in[10];
    const float* Wo     = (const float*)d_in[11];
    const float* bo     = (const float*)d_in[12];
    float* out          = (float*)d_out;

    _Float16* Wcq = (_Float16*)d_ws;
    _Float16* Wck = Wcq + (size_t)WSTRIDE;
    _Float16* Wcv = Wck + (size_t)WSTRIDE;
    _Float16* Woc = Wcv + (size_t)WSTRIDE;

    const int B = in_sizes[0] / (N_NODES * D_MODEL);

    hipLaunchKernelGGL(convert_weights, dim3(1024), dim3(256), 0, stream,
                       Wq, Wk, Wv, Wo, Wcq, Wck, Wcv, Woc);
    hipLaunchKernelGGL(attn_layer_kernel, dim3(B), dim3(256), 0, stream,
                       x, Wcq,
                       lnq_g, lnq_b, lnk_g, lnk_b, lnv_g, lnv_b,
                       rel_t, Woc, bo, out);
}
